// TATransEModel_11630771437658
// MI455X (gfx1250) — matrix operand order repaired and checked by reading, not compiled, not run on hardware
//
#include <hip/hip_runtime.h>
#include <hip/hip_bf16.h>

typedef __attribute__((ext_vector_type(16))) _Float16 v16h;
typedef __attribute__((ext_vector_type(8)))  _Float16 v8h;
typedef __attribute__((ext_vector_type(4)))  _Float16 v4h;
typedef __attribute__((ext_vector_type(8)))  float    v8f;
typedef unsigned int v4u __attribute__((ext_vector_type(4)));
typedef int          v8i __attribute__((ext_vector_type(8)));
typedef int          v4i __attribute__((ext_vector_type(4)));

#define DDIM 128            // embedding / hidden dim
#define NG   512            // 4*D gate rows
#define KTOT 256            // D (x) + D (h)
#define TLEN 5              // rel token + 4 temporal tokens
#define ROWS_PER_WAVE 16
#define WAVES_PER_BLOCK 4
#define ROWS_PER_BLOCK (ROWS_PER_WAVE * WAVES_PER_BLOCK)  // 64
#define BATCH 4096

// LDS layout (dynamic blob, starts at LDS offset 0):
#define SW_BYTES   (NG * KTOT * 2)                 // 262144 : Wc f16 [512][256]
#define SX_BYTES   (ROWS_PER_BLOCK * DDIM * 2)     // 16384  : x_t  f16 [64][128]
#define SH_BYTES   (ROWS_PER_BLOCK * DDIM * 2)     // 16384  : h    f16 [64][128] (x2)
#define SB_BYTES   (NG * 4)                        // 2048   : bias f32 [512]
#define SX_OFF     (SW_BYTES)
#define SH0_OFF    (SX_OFF + SX_BYTES)
#define SH1_OFF    (SH0_OFF + SH_BYTES)
#define SB_OFF     (SH1_OFF + SH_BYTES)
#define SMEM_TOTAL (SB_OFF + SB_BYTES)             // 313344 B <= 320 KB/WGP

#if defined(__has_builtin)
#if __has_builtin(__builtin_amdgcn_tensor_load_to_lds)
#define USE_TDM 1
#endif
#if __has_builtin(__builtin_amdgcn_global_load_async_to_lds_b128)
#define USE_ASYNC_LDS 1
#endif
#endif

// -------- prep: repack [W_ih;W_hh] (f32 [512][128] each) into Wc f16 with
// gate-interleaved rows: n_new = ((n & 127) << 2) | (n >> 7), i.e. the four
// gate rows (i,f,g,o) of one output column sit in 4 adjacent 256-half rows.
// This keeps all B-fragment DS accesses for one (j,laneRow) within a 2 KB
// window -> single base VGPR + immediate offsets (16-bit DS offset field).
// bias = b_ih + b_hh (original gate order). --------
__global__ void tatranse_prep_kernel(const float* __restrict__ W_ih,
                                     const float* __restrict__ W_hh,
                                     const float* __restrict__ b_ih,
                                     const float* __restrict__ b_hh,
                                     _Float16* __restrict__ Wc,
                                     float* __restrict__ bias) {
  int t = blockIdx.x * blockDim.x + threadIdx.x;
  if (t < NG * KTOT) {
    int n = t >> 8;                     // original gate-row index (0..511)
    int k = t & (KTOT - 1);
    float v = (k < DDIM) ? W_ih[n * DDIM + k] : W_hh[n * DDIM + (k - DDIM)];
    int n_new = ((n & 127) << 2) | (n >> 7);
    Wc[n_new * KTOT + k] = (_Float16)v;
  }
  if (t < NG) bias[t] = b_ih[t] + b_hh[t];
}

__device__ __forceinline__ float fast_sigmoid(float x) {
  return 1.0f / (1.0f + __expf(-x));
}

// -------- main: per-block LSTM over 64 batch rows + L1 scoring --------
__attribute__((amdgpu_num_vgpr(256)))
__global__ void __launch_bounds__(WAVES_PER_BLOCK * 32)
tatranse_lstm_score_kernel(
    const int* __restrict__ pos_h, const int* __restrict__ pos_t,
    const int* __restrict__ pos_r, const int* __restrict__ pos_tem,
    const int* __restrict__ neg_h, const int* __restrict__ neg_t,
    const float* __restrict__ ent_emb, const float* __restrict__ rel_emb,
    const float* __restrict__ tem_emb,
    const _Float16* __restrict__ Wc, const float* __restrict__ bias,
    float* __restrict__ out) {
  extern __shared__ char smem[];
  _Float16* sW  = (_Float16*)smem;                 // [512][256] f16 (interleaved)
  _Float16* sX  = (_Float16*)(smem + SX_OFF);      // [64][128] f16
  _Float16* sH0 = (_Float16*)(smem + SH0_OFF);     // [64][128] f16
  _Float16* sH1 = (_Float16*)(smem + SH1_OFF);     // [64][128] f16
  float*    sB  = (float*)(smem + SB_OFF);         // [512] f32

  const int tid      = threadIdx.x;
  const int lane     = tid & 31;
  const int wave     = tid >> 5;
  const int laneRow  = lane & 15;   // N / M position inside a 16-wide tile
  const int laneHalf = lane >> 4;   // which half of the k-chunk this lane holds
  const int blockRow0 = blockIdx.x * ROWS_PER_BLOCK;

  // ---- stage gate weights into LDS ----
#ifdef USE_TDM
  if (wave == 0) {
    const unsigned long long ga = (unsigned long long)(const void*)Wc;
    // D# group 0: count=1 (user), lds_addr=0 (blob base), global_addr, type=2
    v4u g0 = {1u,
              0u,
              (unsigned)(ga & 0xFFFFFFFFu),
              (unsigned)((ga >> 32) & 0x1FFFFFFu) | (2u << 30)};
    // D# group 1: data_size=2B; tensor 256x512 halves, stride0=256; tile = whole
    v8i g1 = {(int)(1u << 16),        // workgroup_mask=0, data_size=1 (2 bytes)
              (int)(256u << 16),      // tensor_dim0[15:0]=256 in bits[63:48]
              (int)(512u << 16),      // tensor_dim0 hi=0 | tensor_dim1 lo=512
              (int)(256u << 16),      // tensor_dim1 hi=0 | tile_dim0=256
              512,                    // tile_dim1=512, tile_dim2=0
              256,                    // tensor_dim0_stride lo32 = 256
              0, 0};
    v4i z4 = {0, 0, 0, 0};
#if __clang_major__ >= 23
    v8i z8 = {0, 0, 0, 0, 0, 0, 0, 0};
    __builtin_amdgcn_tensor_load_to_lds(g0, g1, z4, z4, z8, 0);
#else
    __builtin_amdgcn_tensor_load_to_lds(g0, g1, z4, z4, 0);
#endif
#if __has_builtin(__builtin_amdgcn_s_wait_tensorcnt)
    __builtin_amdgcn_s_wait_tensorcnt(0);
#else
    asm volatile("s_wait_tensorcnt 0x0" ::: "memory");
#endif
  }
#elif defined(USE_ASYNC_LDS)
  {
    const char* srcb = (const char*)Wc;
    const int n16 = SW_BYTES / 16;
    for (int i = tid; i < n16; i += blockDim.x) {
      __builtin_amdgcn_global_load_async_to_lds_b128(
          (__attribute__((address_space(1))) void*)(srcb + i * 16),
          (__attribute__((address_space(3))) void*)((char*)sW + i * 16), 0, 0);
    }
#if __has_builtin(__builtin_amdgcn_s_wait_asynccnt)
    __builtin_amdgcn_s_wait_asynccnt(0);
#else
    asm volatile("s_wait_asynccnt 0x0" ::: "memory");
#endif
  }
#else
  {
    const uint4* src = (const uint4*)Wc;
    uint4* dst = (uint4*)sW;
    const int n16 = SW_BYTES / 16;
    for (int i = tid; i < n16; i += blockDim.x) dst[i] = src[i];
  }
#endif

  // ---- stage bias into LDS; zero initial hidden state buffer sH0 ----
  for (int i = tid; i < NG; i += blockDim.x) sB[i] = bias[i];
  {
    v8h z;
#pragma unroll
    for (int e = 0; e < 8; ++e) z[e] = (_Float16)0.0f;
    const int nchunks = ROWS_PER_BLOCK * (DDIM / 8);
    for (int i = tid; i < nchunks; i += blockDim.x) *((v8h*)sH0 + i) = z;
  }
  __syncthreads();

  // this wave's row slabs
  _Float16* xsl  = sX  + wave * ROWS_PER_WAVE * DDIM;
  _Float16* h0sl = sH0 + wave * ROWS_PER_WAVE * DDIM;
  _Float16* h1sl = sH1 + wave * ROWS_PER_WAVE * DDIM;

  // per-lane A-fragment base pointers (immediate kc offsets from here)
  const _Float16* axbase = xsl  + laneRow * DDIM + laneHalf * 8;
  const _Float16* ah0    = h0sl + laneRow * DDIM + laneHalf * 8;
  const _Float16* ah1    = h1sl + laneRow * DDIM + laneHalf * 8;

  // persistent cell state in WMMA C/D register layout: 8 tiles of v8f
  v8f cfrag[8];
#pragma unroll
  for (int j = 0; j < 8; ++j)
#pragma unroll
    for (int e = 0; e < 8; ++e) cfrag[j][e] = 0.0f;

  for (int t = 0; t < TLEN; ++t) {
    // ---- stage x_t (f32 -> f16): 2 lanes per row, 64 floats each ----
    {
      const int gm  = blockRow0 + wave * ROWS_PER_WAVE + laneRow;
      const int idx = (t == 0) ? pos_r[gm] : pos_tem[gm * 4 + (t - 1)];
      const float* src = ((t == 0) ? rel_emb : tem_emb) + (size_t)idx * DDIM;
      const int cbase = laneHalf * 64;
#pragma unroll
      for (int q = 0; q < 16; ++q) {
        const float4 f = ((const float4*)(src + cbase))[q];
        v4h hv;
        hv[0] = (_Float16)f.x; hv[1] = (_Float16)f.y;
        hv[2] = (_Float16)f.z; hv[3] = (_Float16)f.w;
        *(v4h*)(xsl + laneRow * DDIM + cbase + q * 4) = hv;
      }
    }
    __syncthreads();

    const _Float16* ahcur = (t & 1) ? ah1 : ah0;   // read h_t
    _Float16*       hnxt  = (t & 1) ? h0sl : h1sl; // write h_{t+1}

    // ---- gates = [x,h] @ Wc^T + b; cell update, 16 gate-columns at a time ----
#pragma unroll
    for (int j = 0; j < 8; ++j) {
      // single B base per j: rows [j*16+laneRow][g] are 4 adjacent 256-half
      // rows; g -> +512B, kc -> +64B, laneHalf -> +16B (all DS immediates)
      const _Float16* bbase =
          sW + (((j * 16 + laneRow) << 2) * KTOT) + laneHalf * 8;

      v8f acc[4];
#pragma unroll
      for (int g = 0; g < 4; ++g)
#pragma unroll
        for (int e = 0; e < 8; ++e) acc[g][e] = 0.0f;

      for (int kc = 0; kc < 8; ++kc) {
        // A fragment: rows of [x_t | h_t] for this k-chunk
        const _Float16* ap = ((kc < 4) ? axbase : ahcur) + (kc & 3) * 32;
        v8h alo = *(const v8h*)ap;
        v8h ahi = *(const v8h*)(ap + 16);
        v16h af = __builtin_shufflevector(alo, ahi, 0, 1, 2, 3, 4, 5, 6, 7,
                                          8, 9, 10, 11, 12, 13, 14, 15);
#pragma unroll
        for (int g = 0; g < 4; ++g) {
          const _Float16* bp = bbase + g * KTOT + kc * 32;
          v8h blo = *(const v8h*)bp;
          v8h bhi = *(const v8h*)(bp + 16);
          v16h bf = __builtin_shufflevector(blo, bhi, 0, 1, 2, 3, 4, 5, 6, 7,
                                            8, 9, 10, 11, 12, 13, 14, 15);
          acc[g] = __builtin_amdgcn_wmma_f32_16x16x32_f16(
              false, af, false, bf, (short)0, acc[g], false, false);
        }
      }

      // epilogue: add bias, apply nonlinearities, update c, emit h_{t+1}
      const float b0 = sB[0 * DDIM + j * 16 + laneRow];
      const float b1 = sB[1 * DDIM + j * 16 + laneRow];
      const float b2 = sB[2 * DDIM + j * 16 + laneRow];
      const float b3 = sB[3 * DDIM + j * 16 + laneRow];
#pragma unroll
      for (int e = 0; e < 8; ++e) {
        const float iv = fast_sigmoid(acc[0][e] + b0);
        const float fv = fast_sigmoid(acc[1][e] + b1);
        const float gv = tanhf(acc[2][e] + b2);
        const float ov = fast_sigmoid(acc[3][e] + b3);
        const float cv = fv * cfrag[j][e] + iv * gv;
        cfrag[j][e] = cv;
        const float hv = ov * tanhf(cv);
        const int m = e + 8 * laneHalf;  // row inside wave slab (C/D layout)
        hnxt[m * DDIM + j * 16 + laneRow] = (_Float16)hv;
      }
    }
    __syncthreads();
  }

  // ---- L1 scoring: one thread per batch row; final h lives in sH1 (t=4) ----
  if (tid < ROWS_PER_BLOCK) {
    const int gm = blockRow0 + tid;
    const _Float16* hrow = sH1 + tid * DDIM;  // TLEN=5 -> last write hit buf 1
    const float* ph = ent_emb + (size_t)pos_h[gm] * DDIM;
    const float* pt = ent_emb + (size_t)pos_t[gm] * DDIM;
    const float* nh = ent_emb + (size_t)neg_h[gm] * DDIM;
    const float* nt = ent_emb + (size_t)neg_t[gm] * DDIM;
    float accp = 0.0f, accn = 0.0f;
    for (int k = 0; k < DDIM; k += 4) {
      const float4 a = *(const float4*)(ph + k);
      const float4 b = *(const float4*)(pt + k);
      const float4 c = *(const float4*)(nh + k);
      const float4 d = *(const float4*)(nt + k);
      const v4h hh = *(const v4h*)(hrow + k);
      const float h0 = (float)hh[0], h1 = (float)hh[1];
      const float h2 = (float)hh[2], h3 = (float)hh[3];
      accp += fabsf(a.x + h0 - b.x) + fabsf(a.y + h1 - b.y) +
              fabsf(a.z + h2 - b.z) + fabsf(a.w + h3 - b.w);
      accn += fabsf(c.x + h0 - d.x) + fabsf(c.y + h1 - d.y) +
              fabsf(c.z + h2 - d.z) + fabsf(c.w + h3 - d.w);
    }
    out[gm]         = accp;
    out[BATCH + gm] = accn;
  }
}

extern "C" void kernel_launch(void* const* d_in, const int* in_sizes, int n_in,
                              void* d_out, int out_size, void* d_ws, size_t ws_size,
                              hipStream_t stream) {
  const int* pos_h   = (const int*)d_in[0];
  const int* pos_t   = (const int*)d_in[1];
  const int* pos_r   = (const int*)d_in[2];
  const int* pos_tem = (const int*)d_in[3];
  const int* neg_h   = (const int*)d_in[4];
  const int* neg_t   = (const int*)d_in[5];
  // d_in[6] (neg_r) and d_in[7] (neg_tem) are dead in the reference scoring.
  const float* ent_emb = (const float*)d_in[8];
  const float* rel_emb = (const float*)d_in[9];
  const float* tem_emb = (const float*)d_in[10];
  const float* W_ih = (const float*)d_in[11];
  const float* W_hh = (const float*)d_in[12];
  const float* b_ih = (const float*)d_in[13];
  const float* b_hh = (const float*)d_in[14];

  _Float16* Wc = (_Float16*)d_ws;                          // 512*256 f16
  float* bias  = (float*)((char*)d_ws + (size_t)SW_BYTES); // 512 f32

  tatranse_prep_kernel<<<(NG * KTOT + 255) / 256, 256, 0, stream>>>(
      W_ih, W_hh, b_ih, b_hh, Wc, bias);

  (void)hipFuncSetAttribute((const void*)tatranse_lstm_score_kernel,
                            hipFuncAttributeMaxDynamicSharedMemorySize,
                            (int)SMEM_TOTAL);
  tatranse_lstm_score_kernel<<<BATCH / ROWS_PER_BLOCK, WAVES_PER_BLOCK * 32,
                               SMEM_TOTAL, stream>>>(
      pos_h, pos_t, pos_r, pos_tem, neg_h, neg_t,
      ent_emb, rel_emb, tem_emb, Wc, bias, (float*)d_out);
}
